// ModernConvEqProp_40338332844302
// MI455X (gfx1250) — matrix operand order repaired and checked by reading, not compiled, not run on hardware
//
#include <hip/hip_runtime.h>
#include <hip/hip_bf16.h>

// ---------------------------------------------------------------------------
// ModernConvEqProp on MI455X (gfx1250): f16 WMMA implicit-GEMM convolutions.
// ---------------------------------------------------------------------------

typedef __attribute__((ext_vector_type(16))) _Float16 v16h;
typedef __attribute__((ext_vector_type(8)))  float    v8f;
typedef __attribute__((ext_vector_type(4)))  int      v4i;

union AFrag {
    v16h v;
    uint4 q[2];
};

#if __has_builtin(__builtin_amdgcn_global_load_async_to_lds_b128)
#define CDNA5_ASYNC_LDS 1
typedef __attribute__((address_space(1))) v4i* glb_v4i_ptr;
typedef __attribute__((address_space(3))) v4i* lds_v4i_ptr;
#else
#define CDNA5_ASYNC_LDS 0
#endif

// Stage one 16B quad of the B tile: async DMA to LDS when available.
__device__ __forceinline__ void stage_b_quad(const _Float16* __restrict__ gsrc,
                                             _Float16* ldst)
{
#if CDNA5_ASYNC_LDS
    __builtin_amdgcn_global_load_async_to_lds_b128(
        (glb_v4i_ptr)(v4i*)const_cast<_Float16*>(gsrc),
        (lds_v4i_ptr)(v4i*)ldst,
        0, 0);
#else
    *(uint4*)ldst = *(const uint4*)gsrc;
#endif
}

__device__ __forceinline__ void async_join()
{
#if CDNA5_ASYNC_LDS && __has_builtin(__builtin_amdgcn_s_wait_asynccnt)
    __builtin_amdgcn_s_wait_asynccnt(0);
#endif
}

// ---------------------------------------------------------------------------
// Spectral norm: sigma = || W * (W^T u / ||W^T u||) ||  (one power iteration)
// One block, dynamic LDS holds v[k2] followed by red[256].
// ---------------------------------------------------------------------------
__global__ __launch_bounds__(256) void spectral_sigma_kernel(
    const float* __restrict__ W, const float* __restrict__ u,
    float* __restrict__ sigma, int nout, int k2)
{
    extern __shared__ float sh[];
    float* v   = sh;
    float* red = sh + k2;
    const int tid = threadIdx.x;

    for (int j = tid; j < k2; j += 256) {
        float a = 0.f;
        for (int i = 0; i < nout; ++i) a += W[(size_t)i * k2 + j] * u[i];
        v[j] = a;
    }
    __syncthreads();

    float s = 0.f;
    for (int j = tid; j < k2; j += 256) s += v[j] * v[j];
    red[tid] = s;
    __syncthreads();
    for (int st = 128; st > 0; st >>= 1) {
        if (tid < st) red[tid] += red[tid + st];
        __syncthreads();
    }
    const float nv = sqrtf(red[0]) + 1e-12f;
    __syncthreads();

    float s2 = 0.f;
    for (int i = tid; i < nout; i += 256) {
        float r = 0.f;
        for (int j = 0; j < k2; ++j) r += W[(size_t)i * k2 + j] * v[j];
        r /= nv;
        s2 += r * r;
    }
    red[tid] = s2;
    __syncthreads();
    for (int st = 128; st > 0; st >>= 1) {
        if (tid < st) red[tid] += red[tid + st];
        __syncthreads();
    }
    if (tid == 0) {
        float n2 = red[0];
        sigma[0] = n2 / (sqrtf(n2) + 1e-12f);
    }
}

// ---------------------------------------------------------------------------
// Repack OIHW f32 weights -> f16 [kchunk][COUT][32] for WMMA B tiles.
// k = (dy*3+dx)*CIN + ci
// ---------------------------------------------------------------------------
__global__ __launch_bounds__(256) void repack_kernel(
    const float* __restrict__ w, const float* __restrict__ sigma,
    _Float16* __restrict__ wrep, int CIN, int COUT)
{
    const int K = CIN * 9;
    const int total = K * COUT;
    const int i = blockIdx.x * 256 + threadIdx.x;
    if (i >= total) return;
    const int co = i % COUT;
    const int k  = i / COUT;
    const int tap = k / CIN;
    const int ci  = k % CIN;
    const int dy = tap / 3, dx = tap % 3;
    const float val = w[(((size_t)co * CIN + ci) * 3 + dy) * 3 + dx] / sigma[0];
    wrep[((size_t)(k >> 5) * COUT + co) * 32 + (k & 31)] = (_Float16)val;
}

// ---------------------------------------------------------------------------
// Conv1 direct (3->64, 3x3, stride1), NCHW f32 in -> NHWC f32 out. Tiny FLOPs.
// ---------------------------------------------------------------------------
__global__ __launch_bounds__(256) void conv1_direct_kernel(
    const float* __restrict__ x, const float* __restrict__ w1,
    const float* __restrict__ b1, const float* __restrict__ sigma,
    float* __restrict__ out)
{
    __shared__ float ws[64 * 27];
    for (int i = threadIdx.x; i < 64 * 27; i += 256) ws[i] = w1[i] / sigma[0];
    __syncthreads();

    const size_t idx = (size_t)blockIdx.x * 256 + threadIdx.x;  // B*32*32*64, NHWC
    const int c  = (int)(idx & 63);
    const size_t m = idx >> 6;
    const int ox = (int)(m & 31);
    const int oy = (int)((m >> 5) & 31);
    const int b  = (int)(m >> 10);

    float acc = b1[c];
    #pragma unroll
    for (int ci = 0; ci < 3; ++ci) {
        #pragma unroll
        for (int dy = 0; dy < 3; ++dy) {
            const int iy = oy + dy - 1;
            if (iy < 0 || iy >= 32) continue;
            #pragma unroll
            for (int dx = 0; dx < 3; ++dx) {
                const int ix = ox + dx - 1;
                if (ix < 0 || ix >= 32) continue;
                acc += x[(((size_t)b * 3 + ci) * 32 + iy) * 32 + ix] *
                       ws[c * 27 + ci * 9 + dy * 3 + dx];
            }
        }
    }
    out[idx] = acc;
}

// ---------------------------------------------------------------------------
// GroupNorm stats: per (batch, group) mean & rsqrt(var+eps). NHWC f32 input.
// ---------------------------------------------------------------------------
__global__ __launch_bounds__(256) void gn_stats_kernel(
    const float* __restrict__ x, float* __restrict__ mean, float* __restrict__ rvar,
    int HW, int C, int groups)
{
    __shared__ float rs[256], rs2[256];
    const int bg = blockIdx.x;
    const int b = bg / groups, g = bg % groups;
    const int gc = C / groups;
    const int n = HW * gc;
    float s = 0.f, s2 = 0.f;
    for (int i = threadIdx.x; i < n; i += 256) {
        const int p = i / gc, c = g * gc + (i % gc);
        const float v = x[((size_t)b * HW + p) * C + c];
        s += v; s2 += v * v;
    }
    rs[threadIdx.x] = s; rs2[threadIdx.x] = s2;
    __syncthreads();
    for (int st = 128; st > 0; st >>= 1) {
        if (threadIdx.x < st) {
            rs[threadIdx.x]  += rs[threadIdx.x + st];
            rs2[threadIdx.x] += rs2[threadIdx.x + st];
        }
        __syncthreads();
    }
    if (threadIdx.x == 0) {
        const float m = rs[0] / (float)n;
        const float var = rs2[0] / (float)n - m * m;
        mean[bg] = m;
        rvar[bg] = rsqrtf(var + 1e-5f);
    }
}

// ---------------------------------------------------------------------------
// GroupNorm apply (+ optional tanh), writes f16 and/or f32 NHWC.
// ---------------------------------------------------------------------------
template<bool TANH, bool W16, bool W32>
__global__ __launch_bounds__(256) void gn_apply_kernel(
    const float* __restrict__ x, const float* __restrict__ mean,
    const float* __restrict__ rvar, const float* __restrict__ gw,
    const float* __restrict__ gb, _Float16* __restrict__ o16,
    float* __restrict__ o32, int HW, int C, int groups, size_t total)
{
    const size_t i = (size_t)blockIdx.x * 256 + threadIdx.x;
    if (i >= total) return;
    const int c = (int)(i % C);
    const size_t bp = i / C;
    const int b = (int)(bp / HW);
    const int g = c / (C / groups);
    float v = (x[i] - mean[b * groups + g]) * rvar[b * groups + g] * gw[c] + gb[c];
    if (TANH) v = tanhf(v);
    if (W16) o16[i] = (_Float16)v;
    if (W32) o32[i] = v;
}

// ---------------------------------------------------------------------------
// Implicit-GEMM 3x3 conv via WMMA f16 -> f32.
//   act  : NHWC f16 [B, HIN, WIN, CIN]
//   wrep : f16 [K/32][COUT][32] repacked weights (K = 9*CIN)
//   out  : NHWC f32 [B, HOUT, WOUT, COUT] (bias added)
// Block: 256 threads = 8 waves. Block tile = 128 M x 64 N.
// Each wave: 16 M x 64 N = 4 WMMA accumulators; one A fragment feeds 4 WMMAs.
// B tile (64 x 32 halfs = 4KB/chunk) staged to LDS: one b128 per thread,
// asynchronously (GLOBAL_LOAD_ASYNC_TO_LDS_B128 / ASYNCcnt) when available.
// Grid: (M/128, COUT/64). K consumed in 32-chunks, double-buffered LDS.
// ---------------------------------------------------------------------------
template<int CIN, int COUT, int HIN, int WIN, int HOUT, int WOUT, int STRIDE>
__global__ __launch_bounds__(256) void conv_wmma_kernel(
    const _Float16* __restrict__ act, const _Float16* __restrict__ wrep,
    const float* __restrict__ bias, float* __restrict__ out)
{
    constexpr int K   = CIN * 9;
    constexpr int KCH = K / 32;
    constexpr int P   = HOUT * WOUT;

    __shared__ _Float16 Bs[2][64 * 32];   // [n][k] rows, 64B each; double buffered

    const int tid  = threadIdx.x;
    const int wave = tid >> 5;
    const int lane = tid & 31;
    const int hh   = lane >> 4;          // half-wave select (K split per ISA layout)
    const int nl   = lane & 15;

    const int m0 = blockIdx.x * 128 + wave * 16;
    const int n0 = blockIdx.y * 64;

    // Decode this lane's A row (M = b*P + oy*WOUT + ox)
    const int mrow = m0 + nl;
    const int b    = mrow / P;
    const int rem  = mrow % P;
    const int oy   = rem / WOUT;
    const int ox   = rem % WOUT;

    v8f acc[4] = {{}, {}, {}, {}};

    // Stage B tile for chunk 0: 64 rows x 32 halfs = 4KB, one b128 per thread.
    stage_b_quad(wrep + ((size_t)0 * COUT + n0) * 32 + tid * 8, &Bs[0][tid * 8]);
    async_join();
    __syncthreads();

    for (int kt = 0; kt < KCH; ++kt) {
        const int buf = kt & 1;
        // Stage next B tile into the other LDS buffer (overlaps with compute).
        if (kt + 1 < KCH) {
            stage_b_quad(wrep + ((size_t)(kt + 1) * COUT + n0) * 32 + tid * 8,
                         &Bs[buf ^ 1][tid * 8]);
        }

        // ---- A fragment (ISA 16-bit A 16x32 layout):
        //   lanes 0-15 hold K = {0..7, 16..23}; lanes 16-31 hold K = {8..15, 24..31}
        const int tap    = (kt * 32) / CIN;   // fixed within chunk (CIN % 32 == 0)
        const int ciBase = (kt * 32) % CIN;
        const int iy = oy * STRIDE + tap / 3 - 1;
        const int ix = ox * STRIDE + tap % 3 - 1;
        AFrag a;
        if (iy >= 0 && iy < HIN && ix >= 0 && ix < WIN) {
            const _Float16* ap =
                act + (((size_t)b * HIN + iy) * WIN + ix) * CIN + ciBase;
            a.q[0] = *(const uint4*)(ap + hh * 8);          // K = hh*8 .. +7
            a.q[1] = *(const uint4*)(ap + 16 + hh * 8);     // K = 16+hh*8 .. +7
        } else {
            a.q[0] = make_uint4(0u, 0u, 0u, 0u);
            a.q[1] = make_uint4(0u, 0u, 0u, 0u);
        }

        // Prefetch next chunk's A cacheline (global_prefetch_b8, speculative).
        if (kt + 1 < KCH) {
            const int tap2 = ((kt + 1) * 32) / CIN;
            const int ci2  = ((kt + 1) * 32) % CIN;
            const int iy2 = oy * STRIDE + tap2 / 3 - 1;
            const int ix2 = ox * STRIDE + tap2 % 3 - 1;
            if (iy2 >= 0 && iy2 < HIN && ix2 >= 0 && ix2 < WIN)
                __builtin_prefetch(
                    act + (((size_t)b * HIN + iy2) * WIN + ix2) * CIN + ci2, 0, 0);
        }

        // ---- 4 WMMAs share this A fragment across the 64-wide N tile.
        #pragma unroll
        for (int ns = 0; ns < 4; ++ns) {
            AFrag bm;   // lane holds column n = ns*16+nl, K = {hh*8..+7, 16+hh*8..+7}
            const _Float16* bp = &Bs[buf][(ns * 16 + nl) * 32 + hh * 8];
            bm.q[0] = *(const uint4*)(bp);
            bm.q[1] = *(const uint4*)(bp + 16);
            acc[ns] = __builtin_amdgcn_wmma_f32_16x16x32_f16(
                /*neg_a=*/false, a.v, /*neg_b=*/false, bm.v,
                /*c_mod=*/(short)0, acc[ns], /*reuse_a=*/false, /*reuse_b=*/false);
        }

        async_join();       // this wave's async B-stage landed in LDS
        __syncthreads();    // whole block done before buffer swap
    }

    // C/D layout: VGPR r -> M = r + 8*hh, N = nl. Output NHWC: out[M*COUT + n].
    #pragma unroll
    for (int ns = 0; ns < 4; ++ns) {
        const float bv = bias[n0 + ns * 16 + nl];
        #pragma unroll
        for (int r = 0; r < 8; ++r) {
            const int mr = m0 + hh * 8 + r;
            out[(size_t)mr * COUT + n0 + ns * 16 + nl] = acc[ns][r] + bv;
        }
    }
}

// ---------------------------------------------------------------------------
// Equilibrium blend: h = (1-gamma)*h + gamma*tanh(convraw)
// ---------------------------------------------------------------------------
__global__ __launch_bounds__(256) void blend_kernel(
    float* __restrict__ h, const float* __restrict__ cr, size_t total)
{
    const size_t i = (size_t)blockIdx.x * 256 + threadIdx.x;
    if (i >= total) return;
    h[i] = 0.5f * h[i] + 0.5f * tanhf(cr[i]);
}

// ---------------------------------------------------------------------------
// Mean-pool over 8x8 then FC [256 -> 10]. One block per batch element.
// ---------------------------------------------------------------------------
__global__ __launch_bounds__(256) void pool_fc_kernel(
    const float* __restrict__ h, const float* __restrict__ fcw,
    const float* __restrict__ fcb, float* __restrict__ out)
{
    __shared__ float feat[256];
    const int b = blockIdx.x;
    const int c = threadIdx.x;
    float s = 0.f;
    for (int p = 0; p < 64; ++p) s += h[((size_t)b * 64 + p) * 256 + c];
    feat[c] = s * (1.0f / 64.0f);
    __syncthreads();
    if (threadIdx.x < 10) {
        float a = fcb[threadIdx.x];
        for (int cc = 0; cc < 256; ++cc) a += feat[cc] * fcw[threadIdx.x * 256 + cc];
        out[(size_t)b * 10 + threadIdx.x] = a;
    }
}

// ---------------------------------------------------------------------------
// Host orchestration
// ---------------------------------------------------------------------------
extern "C" void kernel_launch(void* const* d_in, const int* in_sizes, int n_in,
                              void* d_out, int out_size, void* d_ws, size_t ws_size,
                              hipStream_t stream) {
    (void)in_sizes; (void)n_in; (void)out_size; (void)ws_size;

    const float* x   = (const float*)d_in[0];
    const float* w1  = (const float*)d_in[1];
    const float* b1  = (const float*)d_in[2];
    const float* u1  = (const float*)d_in[3];
    const float* g1w = (const float*)d_in[4];
    const float* g1b = (const float*)d_in[5];
    const float* w2  = (const float*)d_in[6];
    const float* b2  = (const float*)d_in[7];
    const float* u2  = (const float*)d_in[8];
    const float* g2w = (const float*)d_in[9];
    const float* g2b = (const float*)d_in[10];
    const float* w3  = (const float*)d_in[11];
    const float* b3  = (const float*)d_in[12];
    const float* u3  = (const float*)d_in[13];
    const float* g3w = (const float*)d_in[14];
    const float* g3b = (const float*)d_in[15];
    const float* weq = (const float*)d_in[16];
    const float* beq = (const float*)d_in[17];
    const float* ueq = (const float*)d_in[18];
    const float* gnw = (const float*)d_in[19];
    const float* gnb = (const float*)d_in[20];
    const float* fcw = (const float*)d_in[21];
    const float* fcb = (const float*)d_in[22];

    // ---- workspace carve-up (256B aligned chunks) ----
    char* p = (char*)d_ws;
    auto alloc = [&](size_t bytes) -> void* {
        void* r = (void*)p;
        p += (bytes + 255) & ~(size_t)255;
        return r;
    };
    float*    sigma = (float*)alloc(4 * sizeof(float));            // sigma[0..3]
    float*    mean  = (float*)alloc(512 * 8 * sizeof(float));
    float*    rvar  = (float*)alloc(512 * 8 * sizeof(float));
    _Float16* wrep2 = (_Float16*)alloc((size_t)576  * 128 * 2);
    _Float16* wrep3 = (_Float16*)alloc((size_t)1152 * 256 * 2);
    _Float16* wrepE = (_Float16*)alloc((size_t)2304 * 256 * 2);
    float*    raw   = (float*)alloc((size_t)512 * 1024 * 64 * 4);  // reused raw conv out
    _Float16* a16   = (_Float16*)alloc((size_t)512 * 1024 * 64 * 2); // reused f16 acts
    float*    hbuf  = (float*)alloc((size_t)512 * 64 * 256 * 4);   // persistent h

    // ---- spectral sigmas (4 single-block kernels) ----
    spectral_sigma_kernel<<<1, 256, (27   + 256) * 4, stream>>>(w1,  u1,  sigma + 0,  64,   27);
    spectral_sigma_kernel<<<1, 256, (576  + 256) * 4, stream>>>(w2,  u2,  sigma + 1, 128,  576);
    spectral_sigma_kernel<<<1, 256, (1152 + 256) * 4, stream>>>(w3,  u3,  sigma + 2, 256, 1152);
    spectral_sigma_kernel<<<1, 256, (2304 + 256) * 4, stream>>>(weq, ueq, sigma + 3, 256, 2304);

    // ---- weight repacks to WMMA B layout ----
    repack_kernel<<<(576  * 128 + 255) / 256, 256, 0, stream>>>(w2,  sigma + 1, wrep2,  64, 128);
    repack_kernel<<<(1152 * 256 + 255) / 256, 256, 0, stream>>>(w3,  sigma + 2, wrep3, 128, 256);
    repack_kernel<<<(2304 * 256 + 255) / 256, 256, 0, stream>>>(weq, sigma + 3, wrepE, 256, 256);

    // ---- stage 1: conv1 (direct) + GN + tanh -> f16 NHWC [512,32,32,64] ----
    conv1_direct_kernel<<<131072, 256, 0, stream>>>(x, w1, b1, sigma + 0, raw);
    gn_stats_kernel<<<4096, 256, 0, stream>>>(raw, mean, rvar, 1024, 64, 8);
    gn_apply_kernel<true, true, false><<<131072, 256, 0, stream>>>(
        raw, mean, rvar, g1w, g1b, a16, nullptr, 1024, 64, 8, (size_t)33554432);

    // ---- stage 2: WMMA conv 64->128 s2 + GN + tanh -> f16 [512,16,16,128] ----
    conv_wmma_kernel<64, 128, 32, 32, 16, 16, 2>
        <<<dim3(1024, 2), 256, 0, stream>>>(a16, wrep2, b2, raw);
    gn_stats_kernel<<<4096, 256, 0, stream>>>(raw, mean, rvar, 256, 128, 8);
    gn_apply_kernel<true, true, false><<<65536, 256, 0, stream>>>(
        raw, mean, rvar, g2w, g2b, a16, nullptr, 256, 128, 8, (size_t)16777216);

    // ---- stage 3: WMMA conv 128->256 s2 + GN + tanh -> f32 h [512,8,8,256] ----
    conv_wmma_kernel<128, 256, 16, 16, 8, 8, 2>
        <<<dim3(256, 4), 256, 0, stream>>>(a16, wrep3, b3, raw);
    gn_stats_kernel<<<4096, 256, 0, stream>>>(raw, mean, rvar, 64, 256, 8);
    gn_apply_kernel<true, false, true><<<32768, 256, 0, stream>>>(
        raw, mean, rvar, g3w, g3b, nullptr, hbuf, 64, 256, 8, (size_t)8388608);

    // ---- equilibrium loop: 15 x (GN -> WMMA conv -> tanh-blend) ----
    for (int step = 0; step < 15; ++step) {
        gn_stats_kernel<<<4096, 256, 0, stream>>>(hbuf, mean, rvar, 64, 256, 8);
        gn_apply_kernel<false, true, false><<<32768, 256, 0, stream>>>(
            hbuf, mean, rvar, gnw, gnb, a16, nullptr, 64, 256, 8, (size_t)8388608);
        conv_wmma_kernel<256, 256, 8, 8, 8, 8, 1>
            <<<dim3(256, 4), 256, 0, stream>>>(a16, wrepE, beq, raw);
        blend_kernel<<<32768, 256, 0, stream>>>(hbuf, raw, (size_t)8388608);
    }

    // ---- mean-pool + FC -> logits [512,10] ----
    pool_fc_kernel<<<512, 256, 0, stream>>>(hbuf, fcw, fcb, (float*)d_out);
}
